// GeneratorTransformerEncoderBlock_22076131902044
// MI455X (gfx1250) — compile-verified, matrix operand order checked by reading, not run on hardware
//
#include <hip/hip_runtime.h>
#include <hip/hip_bf16.h>

typedef __attribute__((ext_vector_type(16))) __bf16 bf16x16;
typedef __attribute__((ext_vector_type(8)))  __bf16 bf16x8;
typedef __attribute__((ext_vector_type(8)))  float  v8f;

union TileU { bf16x16 v; bf16x8 h[2]; };

// Load a 16x32 bf16 tile in WMMA A-layout (same mirrored layout used for B)
// from a row-major matrix (global or LDS): lane<16 -> row=base+(lane&15),
// K={0..7,16..23}; lane>=16 -> same row, K={8..15,24..31}.
__device__ __forceinline__ bf16x16 ld_tile16x32(const __bf16* base, int ld,
                                                int row, int kBase, int lane) {
    const __bf16* p = base + (size_t)(row + (lane & 15)) * ld + kBase + ((lane >> 4) << 3);
    TileU u;
    u.h[0] = *(const bf16x8*)(p);
    u.h[1] = *(const bf16x8*)(p + 16);
    return u.v;
}

// Async global -> LDS copy of 16 bytes per lane (CDNA5 ASYNCcnt path).
__device__ __forceinline__ void async_copy_b128(void* lds, const void* g) {
    unsigned loff = (unsigned)(unsigned long long)(__attribute__((address_space(3))) char*)lds;
    asm volatile("global_load_async_to_lds_b128 %0, %1, off"
                 :: "v"(loff), "v"(g) : "memory");
}

// ---------------- small helper kernels ----------------

__global__ void cvt_bf16_kernel(const float* __restrict__ src, __bf16* __restrict__ dst, int n) {
    int i = (blockIdx.x * blockDim.x + threadIdx.x) * 4;
    if (i < n) {
        float4 f = *(const float4*)(src + i);
        dst[i + 0] = (__bf16)f.x; dst[i + 1] = (__bf16)f.y;
        dst[i + 2] = (__bf16)f.z; dst[i + 3] = (__bf16)f.w;
    }
}

// gamma[b,d] = w[b,:]·Wg[d,:] + bg[d]; beta likewise. 2*B*D = 8192 threads.
__global__ void gamma_beta_kernel(const float* __restrict__ w,
                                  const float* __restrict__ Wg, const float* __restrict__ bg,
                                  const float* __restrict__ Wb, const float* __restrict__ bb,
                                  float* __restrict__ gamma, float* __restrict__ beta) {
    int idx = blockIdx.x * blockDim.x + threadIdx.x;   // 0..8191
    int sel = idx >> 12;                               // 0 = gamma, 1 = beta
    int j   = idx & 4095;
    int b   = j >> 10;
    int d   = j & 1023;
    const float* Wrow = (sel ? Wb : Wg) + (size_t)d * 1024;
    const float* wr   = w + b * 1024;
    float acc = 0.0f;
    for (int k = 0; k < 1024; k += 4) {
        float4 a = *(const float4*)(wr + k);
        float4 c = *(const float4*)(Wrow + k);
        acc += a.x * c.x + a.y * c.y + a.z * c.z + a.w * c.w;
    }
    if (sel) beta[j]  = acc + bb[d];
    else     gamma[j] = acc + bg[d];
}

// Per-token self-modulated LayerNorm. mode 0: emit bf16 h. mode 1: emit f32 (sln(x)+x).
__global__ void sln_kernel(const float* __restrict__ x,
                           const float* __restrict__ gamma, const float* __restrict__ beta,
                           const float* __restrict__ ln_g, const float* __restrict__ ln_b,
                           __bf16* __restrict__ out_bf, float* __restrict__ out_f, int mode) {
    int row = blockIdx.x;          // 0..4095 (B*N)
    int b   = row >> 10;
    const float* xr = x + (size_t)row * 1024;
    int t = threadIdx.x;
    float v[4], s = 0.0f, ss = 0.0f;
#pragma unroll
    for (int i = 0; i < 4; ++i) {
        v[i] = xr[t + 256 * i];
        s  += v[i];
        ss += v[i] * v[i];
    }
#pragma unroll
    for (int off = 16; off; off >>= 1) {
        s  += __shfl_down(s,  off, 32);
        ss += __shfl_down(ss, off, 32);
    }
    __shared__ float sm[8], sm2[8];
    int wv = t >> 5, ln = t & 31;
    if (ln == 0) { sm[wv] = s; sm2[wv] = ss; }
    __syncthreads();
    if (t == 0) {
        float a = 0.0f, q = 0.0f;
        for (int i = 0; i < 8; ++i) { a += sm[i]; q += sm2[i]; }
        sm[0] = a; sm2[0] = q;
    }
    __syncthreads();
    float mu   = sm[0]  * (1.0f / 1024.0f);
    float var  = sm2[0] * (1.0f / 1024.0f) - mu * mu;
    float rstd = rsqrtf(var + 1e-5f);
#pragma unroll
    for (int i = 0; i < 4; ++i) {
        int d = t + 256 * i;
        float lnv = (v[i] - mu) * rstd * ln_g[d] + ln_b[d];
        float y   = gamma[b * 1024 + d] * lnv + beta[b * 1024 + d];
        if (mode == 0) out_bf[(size_t)row * 1024 + d] = (__bf16)y;
        else           out_f [(size_t)row * 1024 + d] = y + v[i];
    }
}

// ---------------- 128x128-tile GEMM mainloop (4 waves / block) ----------------
// LDS rows padded to 40 elements (80B) -> 16-lane b128 fragment loads are
// bank-conflict-free (20-bank stride mod 64 hits 16 distinct banks).
// Double-buffered async global->LDS staging; 8 async ops/wave/stage.
__device__ __forceinline__ void gemm_mainloop_128(
        const __bf16* __restrict__ A, const __bf16* __restrict__ B,
        int m0, int c0,
        __bf16 (&Alds)[2][128][40], __bf16 (&Blds)[2][128][40],
        v8f (&acc)[4][4], int lane, int wr, int wc) {
    int t = threadIdx.x;   // 0..127
    const __bf16* Ar = A + (size_t)(m0 + t) * 1024;
    const __bf16* Br = B + (size_t)(c0 + t) * 1024;
#pragma unroll
    for (int j = 0; j < 4; ++j) {
        async_copy_b128(&Alds[0][t][j * 8], Ar + j * 8);
        async_copy_b128(&Blds[0][t][j * 8], Br + j * 8);
    }
#pragma unroll 1
    for (int kt = 0; kt < 32; ++kt) {
        int s = kt & 1;
        if (kt + 1 < 32) {
            int k2 = (kt + 1) << 5;
#pragma unroll
            for (int j = 0; j < 4; ++j) {
                async_copy_b128(&Alds[s ^ 1][t][j * 8], Ar + k2 + j * 8);
                async_copy_b128(&Blds[s ^ 1][t][j * 8], Br + k2 + j * 8);
            }
            asm volatile("s_wait_asynccnt 0x8" ::: "memory");
        } else {
            asm volatile("s_wait_asynccnt 0x0" ::: "memory");
        }
        __syncthreads();
        bf16x16 at[4], bt[4];
#pragma unroll
        for (int i = 0; i < 4; ++i) at[i] = ld_tile16x32(&Alds[s][0][0], 40, wr * 64 + i * 16, 0, lane);
#pragma unroll
        for (int i = 0; i < 4; ++i) bt[i] = ld_tile16x32(&Blds[s][0][0], 40, wc * 64 + i * 16, 0, lane);
#pragma unroll
        for (int mi = 0; mi < 4; ++mi)
#pragma unroll
            for (int ni = 0; ni < 4; ++ni)
                acc[mi][ni] = __builtin_amdgcn_wmma_f32_16x16x32_bf16(
                    false, at[mi], false, bt[ni], (short)0, acc[mi][ni], false, false);
        __syncthreads();
    }
}

// ---------------- QKV GEMM ----------------
// qkv[b,n,c] = h[b,n,:]·Wqkv[c,:] + bqkv[c]; c -> (head=c/192, d=(c%192)/3, s=c%3)
// s==0 -> q[B,H,N,hd], s==1 -> k[B,H,N,hd], s==2 -> Vt[B,H,hd,N]
__global__ void __launch_bounds__(128) qkv_gemm_kernel(
        const __bf16* __restrict__ A, const __bf16* __restrict__ W,
        const float* __restrict__ bqkv,
        __bf16* __restrict__ qb, __bf16* __restrict__ kbuf, __bf16* __restrict__ vtb) {
    __shared__ __align__(16) __bf16 Alds[2][128][40];
    __shared__ __align__(16) __bf16 Blds[2][128][40];
    int lane = threadIdx.x & 31, wv = threadIdx.x >> 5;
    int wr = wv >> 1, wc = wv & 1;
    int c0 = blockIdx.x * 128;   // 24 tiles over 3072 channels
    int m0 = blockIdx.y * 128;   // 32 tiles over 4096 tokens
    v8f zero = {};
    v8f acc[4][4];
#pragma unroll
    for (int mi = 0; mi < 4; ++mi)
#pragma unroll
        for (int ni = 0; ni < 4; ++ni) acc[mi][ni] = zero;

    gemm_mainloop_128(A, W, m0, c0, Alds, Blds, acc, lane, wr, wc);

#pragma unroll
    for (int mi = 0; mi < 4; ++mi)
#pragma unroll
        for (int ni = 0; ni < 4; ++ni)
#pragma unroll
            for (int i = 0; i < 8; ++i) {
                int row = m0 + wr * 64 + mi * 16 + ((lane >> 4) << 3) + i;
                int c   = c0 + wc * 64 + ni * 16 + (lane & 15);
                float val = acc[mi][ni][i] + bqkv[c];
                int hh = c / 192;
                int r  = c - hh * 192;
                int d  = r / 3;
                int s  = r - d * 3;
                int bb = row >> 10;
                int n  = row & 1023;
                size_t bh = (size_t)(bb * 16 + hh);
                __bf16 bv = (__bf16)val;
                if (s == 0)      qb  [(bh * 1024 + n) * 64 + d] = bv;
                else if (s == 1) kbuf[(bh * 1024 + n) * 64 + d] = bv;
                else             vtb [(bh * 64 + d) * 1024 + n] = bv;
            }
}

// ---------------- Output projection + bias + residual ----------------
__global__ void __launch_bounds__(128) proj_gemm_kernel(
        const __bf16* __restrict__ A, const __bf16* __restrict__ W,
        const float* __restrict__ bp, const float* __restrict__ hidden,
        float* __restrict__ h1) {
    __shared__ __align__(16) __bf16 Alds[2][128][40];
    __shared__ __align__(16) __bf16 Blds[2][128][40];
    int lane = threadIdx.x & 31, wv = threadIdx.x >> 5;
    int wr = wv >> 1, wc = wv & 1;
    int c0 = blockIdx.x * 128;   // 8 tiles over 1024 channels
    int m0 = blockIdx.y * 128;   // 32 tiles over 4096 tokens
    v8f zero = {};
    v8f acc[4][4];
#pragma unroll
    for (int mi = 0; mi < 4; ++mi)
#pragma unroll
        for (int ni = 0; ni < 4; ++ni) acc[mi][ni] = zero;

    gemm_mainloop_128(A, W, m0, c0, Alds, Blds, acc, lane, wr, wc);

#pragma unroll
    for (int mi = 0; mi < 4; ++mi)
#pragma unroll
        for (int ni = 0; ni < 4; ++ni)
#pragma unroll
            for (int i = 0; i < 8; ++i) {
                int row = m0 + wr * 64 + mi * 16 + ((lane >> 4) << 3) + i;
                int c   = c0 + wc * 64 + ni * 16 + (lane & 15);
                size_t idx = (size_t)row * 1024 + c;
                h1[idx] = acc[mi][ni][i] + bp[c] + hidden[idx];
            }
}

// ---------------- Flash attention ----------------
// 4 waves / block = 4 query tiles of the same (b,h); K (32x64) and Vt (64x32)
// staged in LDS via async copies and shared by all 4 waves (4x reuse).
// S^T = K·Q^T per wave: per-query softmax stats in-lane + one shfl_xor(16);
// exp(S^T) maps element-for-element to the P B-matrix for O^T = Vt·P.
__global__ void __launch_bounds__(128) attn_kernel(
        const __bf16* __restrict__ Q, const __bf16* __restrict__ K,
        const __bf16* __restrict__ Vt, __bf16* __restrict__ Out) {
    __shared__ __align__(16) __bf16 Klds[2][32][72];
    __shared__ __align__(16) __bf16 Vlds[2][64][40];
    int t = threadIdx.x, lane = t & 31, wv = t >> 5;
    int bh = blockIdx.x >> 4;                    // b*16 + h
    int qt = ((blockIdx.x & 15) << 2) + wv;      // query tile 0..63
    const __bf16* q  = Q  + (size_t)bh * 65536;
    const __bf16* kk = K  + (size_t)bh * 65536;
    const __bf16* vt = Vt + (size_t)bh * 65536;
    int n0 = qt * 16;

    bf16x16 qb0 = ld_tile16x32(q, 64, n0, 0, lane);   // B-matrix: hd 0..31 x 16 queries
    bf16x16 qb1 = ld_tile16x32(q, 64, n0, 32, lane);  // hd 32..63
    v8f zero = {};
    v8f o[4] = {zero, zero, zero, zero};              // O^T: 64(hd) x 16(q) f32
    float m_run = -3.0e38f, l_run = 0.0f;

    // stage copy: K tile rows kb..kb+31 (contiguous), Vt tile 64 rows x 32 cols
    auto copyKV = [&](int s, int kb) {
#pragma unroll
        for (int u = 0; u < 2; ++u) {
            int c = t * 2 + u;                        // 0..255
            int kr = c >> 3, kj = c & 7;              // K: 32 rows x 8 chunks
            async_copy_b128(&Klds[s][kr][kj * 8], kk + (size_t)(kb + kr) * 64 + kj * 8);
            int vr = c >> 2, vj = c & 3;              // Vt: 64 rows x 4 chunks
            async_copy_b128(&Vlds[s][vr][vj * 8], vt + (size_t)vr * 1024 + kb + vj * 8);
        }
    };

    copyKV(0, 0);
#pragma unroll 1
    for (int it = 0; it < 32; ++it) {
        int kb = it << 5, s = it & 1;
        if (it + 1 < 32) {
            copyKV(s ^ 1, kb + 32);
            asm volatile("s_wait_asynccnt 0x4" ::: "memory");
        } else {
            asm volatile("s_wait_asynccnt 0x0" ::: "memory");
        }
        __syncthreads();

        const __bf16* Kb = &Klds[s][0][0];
        bf16x16 ka00 = ld_tile16x32(Kb, 72, 0,  0,  lane);
        bf16x16 ka01 = ld_tile16x32(Kb, 72, 0,  32, lane);
        bf16x16 ka10 = ld_tile16x32(Kb, 72, 16, 0,  lane);
        bf16x16 ka11 = ld_tile16x32(Kb, 72, 16, 32, lane);
        v8f s0 = __builtin_amdgcn_wmma_f32_16x16x32_bf16(false, ka00, false, qb0, (short)0, zero, false, false);
        s0     = __builtin_amdgcn_wmma_f32_16x16x32_bf16(false, ka01, false, qb1, (short)0, s0,   false, false);
        v8f s1 = __builtin_amdgcn_wmma_f32_16x16x32_bf16(false, ka10, false, qb0, (short)0, zero, false, false);
        s1     = __builtin_amdgcn_wmma_f32_16x16x32_bf16(false, ka11, false, qb1, (short)0, s1,   false, false);

        float lm = s0[0];
#pragma unroll
        for (int i = 1; i < 8; ++i) lm = fmaxf(lm, s0[i]);
#pragma unroll
        for (int i = 0; i < 8; ++i) lm = fmaxf(lm, s1[i]);
        lm = fmaxf(lm, __shfl_xor(lm, 16, 32));       // other 16 keys of same query
        float m_new = fmaxf(m_run, lm);
        float scale = __expf(m_run - m_new);

        bf16x16 pb;                                    // P as B-matrix: 32(key) x 16(query)
        float ls = 0.0f;
#pragma unroll
        for (int i = 0; i < 8; ++i) { float e = __expf(s0[i] - m_new); ls += e; pb[i]     = (__bf16)e; }
#pragma unroll
        for (int i = 0; i < 8; ++i) { float e = __expf(s1[i] - m_new); ls += e; pb[8 + i] = (__bf16)e; }
        ls += __shfl_xor(ls, 16, 32);
        l_run = l_run * scale + ls;
        m_run = m_new;
#pragma unroll
        for (int t4 = 0; t4 < 4; ++t4)
#pragma unroll
            for (int i = 0; i < 8; ++i) o[t4][i] *= scale;

        const __bf16* Vb = &Vlds[s][0][0];
#pragma unroll
        for (int t4 = 0; t4 < 4; ++t4) {
            bf16x16 va = ld_tile16x32(Vb, 40, t4 * 16, 0, lane);  // Vt: 16(hd) x 32(key)
            o[t4] = __builtin_amdgcn_wmma_f32_16x16x32_bf16(false, va, false, pb, (short)0, o[t4], false, false);
        }
        __syncthreads();
    }

    float inv = 1.0f / (l_run * 32.0f);   // softmax denom * sqrt(D)
    int b = bh >> 4, hh = bh & 15;
    int row = b * 1024 + n0 + (lane & 15);
#pragma unroll
    for (int t4 = 0; t4 < 4; ++t4)
#pragma unroll
        for (int i = 0; i < 8; ++i) {
            int d = hh * 64 + t4 * 16 + ((lane >> 4) << 3) + i;
            Out[(size_t)row * 1024 + d] = (__bf16)(o[t4][i] * inv);
        }
}

// ---------------- orchestration ----------------

extern "C" void kernel_launch(void* const* d_in, const int* in_sizes, int n_in,
                              void* d_out, int out_size, void* d_ws, size_t ws_size,
                              hipStream_t stream) {
    const float* hidden = (const float*)d_in[0];
    const float* w      = (const float*)d_in[1];
    const float* ln_g   = (const float*)d_in[2];
    const float* ln_b   = (const float*)d_in[3];
    const float* Wg     = (const float*)d_in[4];
    const float* bg     = (const float*)d_in[5];
    const float* Wb     = (const float*)d_in[6];
    const float* bb     = (const float*)d_in[7];
    const float* Wqkv   = (const float*)d_in[8];
    const float* bqkv   = (const float*)d_in[9];
    const float* Wp     = (const float*)d_in[10];
    const float* bp     = (const float*)d_in[11];
    // W1/b1/W2/b2 unused (reference discards FFN output); num_heads fixed at 16.

    char* ws = (char*)d_ws;
    float*  gamma   = (float*) (ws + 0);          // 16 KB
    float*  beta    = (float*) (ws + 16384);      // 16 KB
    __bf16* h_bf    = (__bf16*)(ws + 32768);      // 8 MB  [4096,1024]
    __bf16* Wqkv_bf = (__bf16*)(ws + 8421376);    // 6 MB  [3072,1024]
    __bf16* Wp_bf   = (__bf16*)(ws + 14712832);   // 2 MB  [1024,1024]
    __bf16* q_b     = (__bf16*)(ws + 16809984);   // 8 MB  [B,H,N,hd]
    __bf16* k_b     = (__bf16*)(ws + 25198592);   // 8 MB  [B,H,N,hd]
    __bf16* vt_b    = (__bf16*)(ws + 33587200);   // 8 MB  [B,H,hd,N]
    __bf16* attn    = (__bf16*)(ws + 41975808);   // 8 MB  [4096,1024]
    float*  h1      = (float*) (ws + 50364416);   // 16 MB [4096,1024]
    float*  out     = (float*)d_out;

    cvt_bf16_kernel<<<3072 * 1024 / 1024, 256, 0, stream>>>(Wqkv, Wqkv_bf, 3072 * 1024);
    cvt_bf16_kernel<<<1024 * 1024 / 1024, 256, 0, stream>>>(Wp, Wp_bf, 1024 * 1024);
    gamma_beta_kernel<<<32, 256, 0, stream>>>(w, Wg, bg, Wb, bb, gamma, beta);
    sln_kernel<<<4096, 256, 0, stream>>>(hidden, gamma, beta, ln_g, ln_b, h_bf, nullptr, 0);
    qkv_gemm_kernel<<<dim3(24, 32), 128, 0, stream>>>(h_bf, Wqkv_bf, bqkv, q_b, k_b, vt_b);
    attn_kernel<<<1024, 128, 0, stream>>>(q_b, k_b, vt_b, attn);
    proj_gemm_kernel<<<dim3(8, 32), 128, 0, stream>>>(attn, Wp_bf, bp, hidden, h1);
    sln_kernel<<<4096, 256, 0, stream>>>(h1, gamma, beta, ln_g, ln_b, nullptr, out, 1);
}
